// LMulApproxLinearFP16_43602507989104
// MI455X (gfx1250) — matrix operand order repaired and checked by reading, not compiled, not run on hardware
//
#include <hip/hip_runtime.h>
#include <stdint.h>

// L-Mul approximate linear layer for MI455X (gfx1250, wave32).
//
// VALU-bound bit-trick GEMM: packed 16-bit VALU computes signed fp16 lmul
// values; the matrix core accumulates them in fp32 via a constant 0/1
// selector B (B[k',n] = (k'>>1==n)), so one v_wmma_f32_16x16x32_f16 folds
// 512 lmul values into the correct C[m,n] cells (1 wmma per 512 MACs; RNE
// fp32 accumulation like the reference).
//
// Inner chain (confirmed codegen): v_pk_add_u16, v_pk_sub_u16 clamp,
// v_pk_min3_u16 (0x7BFF clamp + both zero-kills), 2x v_xor (sign) =
// 5 packed ops / 2 MACs. This round: K-step 4 dwords so every stream load
// is a B128 (halves vmem+salu issue slots vs B64; VMEM shares issue bw
// with VALU on the SIMD).
//
// Preprocess pass (d_ws): mag = p&0x7FFF7FFF, sgn = p&0x80008000,
// lim = per-half (mag==0 ? 0 : 0x7BFF). 'lim' makes the zero check a min
// with a loaded operand -> fuses into pk_min3, can't be scalarized.

typedef __attribute__((ext_vector_type(16))) _Float16 v16h;
typedef __attribute__((ext_vector_type(8)))  float    v8f;
typedef __attribute__((ext_vector_type(2)))  unsigned short u16x2;

#define LM_M 512
#define LM_K 1024
#define LM_O 1024
#define KD2  (LM_K / 2)   // dword (fp16-pair) units per row
#define KB4  (KD2 / 4)    // uint4 units per row = 128

// d_ws layout in dwords: w streams (1024*512 each), then x streams (512*512 each)
#define W_DW   (LM_O * KD2)          // 524288
#define X_DW   (LM_M * KD2)          // 262144
#define WMAG_OFF 0
#define WSGN_OFF (W_DW)
#define WLIM_OFF (2 * W_DW)
#define XMAG_OFF (3 * W_DW)
#define XSGN_OFF (3 * W_DW + X_DW)
#define XLIM_OFF (3 * W_DW + 2 * X_DW)
#define WS_DWORDS (3 * W_DW + 3 * X_DW)

union V16HU { v16h v; uint32_t w[8]; };

// ---------------- preprocessing: mag / sgn / lim streams -------------------
__global__ __launch_bounds__(256) void lmul_pre_kernel(
    const uint32_t* __restrict__ src, uint32_t* __restrict__ mag,
    uint32_t* __restrict__ sgn, uint32_t* __restrict__ lim, int n)
{
    int i = blockIdx.x * 256 + threadIdx.x;
    if (i >= n) return;
    uint32_t p = src[i];
    uint32_t m = p & 0x7FFF7FFFu;
    mag[i] = m;
    sgn[i] = p & 0x80008000u;
    const u16x2 kOne = {1u, 1u};
    const u16x2 kMax = {0x7BFFu, 0x7BFFu};
    u16x2 mv = __builtin_bit_cast(u16x2, m);
    u16x2 z  = __builtin_elementwise_min(mv, kOne);  // 0/1 per half
    u16x2 l  = z * kMax;                             // 0x0000 or 0x7BFF
    lim[i] = __builtin_bit_cast(uint32_t, l);
}

// ------------- main chain: 5 packed ops per fp16 pair (2 MACs) -------------
static __device__ __forceinline__ uint32_t lmul2_pre(
    uint32_t xm, uint32_t xl, uint32_t xs,
    uint32_t wm, uint32_t wl, uint32_t ws)
{
    const u16x2 kBias = {0x3BC0u, 0x3BC0u};  // 0x3C00 - 64
    u16x2 s = __builtin_bit_cast(u16x2, xm) + __builtin_bit_cast(u16x2, wm);
#if __has_builtin(__builtin_elementwise_sub_sat)
    u16x2 u = __builtin_elementwise_sub_sat(s, kBias);            // clamp low
#else
    u16x2 u = __builtin_elementwise_max(s, kBias) - kBias;
#endif
    u = __builtin_elementwise_min(u, __builtin_bit_cast(u16x2, wl)); // \ fuse to
    u = __builtin_elementwise_min(u, __builtin_bit_cast(u16x2, xl)); // / pk_min3
    return __builtin_bit_cast(uint32_t, u) ^ xs ^ ws;                // sign
}

__global__ __launch_bounds__(256) void lmul_wmma_pre_kernel(
    const uint4* __restrict__ wmag, const uint4* __restrict__ wsgn,
    const uint4* __restrict__ wlim,
    const uint4* __restrict__ xmag, const uint4* __restrict__ xsgn,
    const uint4* __restrict__ xlim,
    const _Float16* __restrict__ bias, _Float16* __restrict__ out)
{
    const int lane = threadIdx.x & 31;
    const int wave = threadIdx.x >> 5;
    const int tile = blockIdx.x * 8 + wave;        // 2048 tiles of 16x16
    const int o_tiles = LM_O / 16;                 // 64
    const int m_base = (tile / o_tiles) * 16;
    const int o_base = (tile % o_tiles) * 16;
    const int n   = lane & 15;
    const int hig = lane >> 4;

    // A 16x32 f16 layout: lane l, VGPR j, half h -> k' ; o(k') = k'>>1
    const int m = m_base + n;
    int wrow[8];
#pragma unroll
    for (int j = 0; j < 8; ++j)
        wrow[j] = (o_base + j + hig * 8) * KB4;
    const int xrow = m * KB4;

    // Selector B[k',n] = (k'>>1 == n); element e (lane l) = B[e+16*hig, n]
    v16h b;
#pragma unroll
    for (int e = 0; e < 16; ++e) {
        int kp = e + 16 * hig;
        b[e] = ((kp >> 1) == n) ? (_Float16)1.0f : (_Float16)0.0f;
    }

    v8f c = {};

    for (int t = 0; t < KB4; ++t) {                // 8 K-elements / 4 WMMAs per iter
        uint4 xm4 = xmag[xrow + t];
        uint4 xs4 = xsgn[xrow + t];
        uint4 xl4 = xlim[xrow + t];
        uint4 wm4[8], ws4[8], wl4[8];
#pragma unroll
        for (int j = 0; j < 8; ++j) {
            wm4[j] = wmag[wrow[j] + t];
            ws4[j] = wsgn[wrow[j] + t];
            wl4[j] = wlim[wrow[j] + t];
        }
        if (t + 4 < KB4) {
            __builtin_prefetch(&xmag[xrow + t + 4], 0, 1);    // global_prefetch_b8
            __builtin_prefetch(&wmag[wrow[0] + t + 4], 0, 1);
        }
#pragma unroll
        for (int s = 0; s < 4; ++s) {
            uint32_t xm = ((const uint32_t*)&xm4)[s];
            uint32_t xs = ((const uint32_t*)&xs4)[s];
            uint32_t xl = ((const uint32_t*)&xl4)[s];
            V16HU a;
#pragma unroll
            for (int j = 0; j < 8; ++j)
                a.w[j] = lmul2_pre(xm, xl, xs,
                                   ((const uint32_t*)&wm4[j])[s],
                                   ((const uint32_t*)&wl4[j])[s],
                                   ((const uint32_t*)&ws4[j])[s]);
            c = __builtin_amdgcn_wmma_f32_16x16x32_f16(
                    false, a.v, false, b, (short)0, c, false, false);
        }
    }

    // D layout: VGPR r, lane l -> M = r + 8*hig, N = n
    const int oc = o_base + n;
    const _Float16 bv = bias[oc];
#pragma unroll
    for (int r = 0; r < 8; ++r) {
        int mr = m_base + r + 8 * hig;
        out[(size_t)mr * LM_O + oc] = (_Float16)c[r] + bv;
    }
}

// ---------------- fallback (no scratch): fused single-pass kernel ----------
static __device__ __forceinline__ uint32_t lmul2_raw(uint32_t xp, uint32_t xm_u, uint32_t wp)
{
    const u16x2 kBias   = {0x3BC0u, 0x3BC0u};
    const u16x2 kMaxFin = {0x7BFFu, 0x7BFFu};
    const u16x2 kOne    = {1u, 1u};
    const u16x2 kZero   = {0u, 0u};
    uint32_t wm_u = wp & 0x7FFF7FFFu;
    u16x2 xm = __builtin_bit_cast(u16x2, xm_u);
    u16x2 wm = __builtin_bit_cast(u16x2, wm_u);
    u16x2 s  = xm + wm;
#if __has_builtin(__builtin_elementwise_sub_sat)
    u16x2 u  = __builtin_elementwise_sub_sat(s, kBias);
#else
    u16x2 u  = __builtin_elementwise_max(s, kBias) - kBias;
#endif
    u = __builtin_elementwise_min(u, kMaxFin);
    u16x2 mn  = __builtin_elementwise_min(xm, wm);
    u16x2 z   = __builtin_elementwise_min(mn, kOne);
    u16x2 msk = kZero - z;
    uint32_t mag = __builtin_bit_cast(uint32_t, u) & __builtin_bit_cast(uint32_t, msk);
    return mag | ((xp ^ wp) & 0x80008000u);
}

__global__ __launch_bounds__(256) void lmul_wmma_kernel(
    const uint32_t* __restrict__ xw, const uint32_t* __restrict__ ww,
    const _Float16* __restrict__ bias, _Float16* __restrict__ out)
{
    const int lane = threadIdx.x & 31;
    const int wave = threadIdx.x >> 5;
    const int tile = blockIdx.x * 8 + wave;
    const int o_tiles = LM_O / 16;
    const int m_base = (tile / o_tiles) * 16;
    const int o_base = (tile % o_tiles) * 16;
    const int n   = lane & 15;
    const int hig = lane >> 4;
    const int m = m_base + n;
    int oj[8];
#pragma unroll
    for (int j = 0; j < 8; ++j)
        oj[j] = o_base + j + hig * 8;
    v16h b;
#pragma unroll
    for (int e = 0; e < 16; ++e) {
        int kp = e + 16 * hig;
        b[e] = ((kp >> 1) == n) ? (_Float16)1.0f : (_Float16)0.0f;
    }
    v8f c = {};
    const uint32_t* xrow = xw + (size_t)m * KD2;
    const uint32_t* wrow[8];
#pragma unroll
    for (int j = 0; j < 8; ++j) wrow[j] = ww + (size_t)oj[j] * KD2;

    for (int kb = 0; kb < KD2; kb += 4) {
        uint4 x4 = *(const uint4*)(xrow + kb);
        uint4 w4[8];
#pragma unroll
        for (int j = 0; j < 8; ++j) w4[j] = *(const uint4*)(wrow[j] + kb);
#pragma unroll
        for (int s = 0; s < 4; ++s) {
            uint32_t xp = ((const uint32_t*)&x4)[s];
            uint32_t xm = xp & 0x7FFF7FFFu;
            V16HU a;
#pragma unroll
            for (int j = 0; j < 8; ++j)
                a.w[j] = lmul2_raw(xp, xm, ((const uint32_t*)&w4[j])[s]);
            c = __builtin_amdgcn_wmma_f32_16x16x32_f16(
                    false, a.v, false, b, (short)0, c, false, false);
        }
    }
    const int oc = o_base + n;
    const _Float16 bv = bias[oc];
#pragma unroll
    for (int r = 0; r < 8; ++r) {
        int mr = m_base + r + 8 * hig;
        out[(size_t)mr * LM_O + oc] = (_Float16)c[r] + bv;
    }
}

extern "C" void kernel_launch(void* const* d_in, const int* in_sizes, int n_in,
                              void* d_out, int out_size, void* d_ws, size_t ws_size,
                              hipStream_t stream) {
    (void)in_sizes; (void)n_in; (void)out_size;
    const uint32_t* x = (const uint32_t*)d_in[0];     // fp16 [512,1024] as dwords
    const uint32_t* w = (const uint32_t*)d_in[1];     // fp16 [1024,1024] as dwords
    const _Float16* bias = (const _Float16*)d_in[2];  // fp16 [1,1024]
    _Float16* out = (_Float16*)d_out;                 // fp16 [512,1024]

    dim3 block(256);
    dim3 grid((LM_M / 16) * (LM_O / 16) / 8);         // 2048 waves, 8 per block

    if (ws_size >= (size_t)WS_DWORDS * 4u) {
        uint32_t* ws = (uint32_t*)d_ws;
        lmul_pre_kernel<<<W_DW / 256, 256, 0, stream>>>(
            w, ws + WMAG_OFF, ws + WSGN_OFF, ws + WLIM_OFF, W_DW);
        lmul_pre_kernel<<<X_DW / 256, 256, 0, stream>>>(
            x, ws + XMAG_OFF, ws + XSGN_OFF, ws + XLIM_OFF, X_DW);
        lmul_wmma_pre_kernel<<<grid, block, 0, stream>>>(
            (const uint4*)(ws + WMAG_OFF), (const uint4*)(ws + WSGN_OFF),
            (const uint4*)(ws + WLIM_OFF),
            (const uint4*)(ws + XMAG_OFF), (const uint4*)(ws + XSGN_OFF),
            (const uint4*)(ws + XLIM_OFF), bias, out);
    } else {
        lmul_wmma_kernel<<<grid, block, 0, stream>>>(x, w, bias, out);
    }
}
